// GemmaAttention_43654047596968
// MI455X (gfx1250) — matrix operand order repaired and checked by reading, not compiled, not run on hardware
//
#include <hip/hip_runtime.h>
#include <hip/hip_bf16.h>
#include <math.h>

typedef __attribute__((ext_vector_type(16))) __bf16 v16bf;
typedef __attribute__((ext_vector_type(8)))  float  v8f;

union Frag {
    v16bf v;
    unsigned int u[8];
};

__device__ __forceinline__ unsigned short f2bf(float f) {
    union { float f; unsigned int u; } x; x.f = f;
    unsigned int r = x.u + 0x7fffu + ((x.u >> 16) & 1u);   // round-to-nearest-even
    return (unsigned short)(r >> 16);
}
__device__ __forceinline__ float bf2f(unsigned short u) {
    union { unsigned int u; float f; } x; x.u = ((unsigned int)u) << 16;
    return x.f;
}
__device__ __forceinline__ unsigned int pack2(float lo, float hi) {
    return (unsigned int)f2bf(lo) | ((unsigned int)f2bf(hi) << 16);
}
// stable tanh: one v_exp_f32 + a few VALU, saturates correctly for large |y|
__device__ __forceinline__ float fast_tanh(float y) {
    float ay = fabsf(y);
    float e  = __expf(ay * 2.0f);
    float t  = 1.0f - 2.0f / (e + 1.0f);
    return copysignf(t, y);
}

// CDNA5 async copy: global -> LDS, 16B per lane, tracked by ASYNCcnt.
__device__ __forceinline__ void async_ld_b128(unsigned lds_off, const void* g) {
    asm volatile("global_load_async_to_lds_b128 %0, %1, off"
                 :: "v"(lds_off), "v"(g) : "memory");
}
__device__ __forceinline__ unsigned lds_addr(const void* p) {
    return (unsigned)(size_t)p;   // addr[31:0] of the LDS aperture = byte offset
}

#define NEG_BIG (-2.3819763e38f)

// ---------------------------------------------------------------------------
// fp32 -> bf16 elementwise convert (vectorized x4)
// ---------------------------------------------------------------------------
__global__ void cvt_bf16_vec4(const float4* __restrict__ in, uint2* __restrict__ out, int n4)
{
    int i = blockIdx.x * blockDim.x + threadIdx.x;
    if (i >= n4) return;
    float4 v = in[i];
    out[i] = make_uint2(pack2(v.x, v.y), pack2(v.z, v.w));
}

// ---------------------------------------------------------------------------
// fp32 W[K][N] -> bf16 Wt[N][K] (tiled transpose through LDS)
// block (32,8), grid (N/32, K/32)
// ---------------------------------------------------------------------------
__global__ __launch_bounds__(256, 1)
void transpose_cvt(const float* __restrict__ W, unsigned short* __restrict__ Wt, int K, int N)
{
    __shared__ float tile[32][33];
    int n0 = blockIdx.x * 32, k0 = blockIdx.y * 32;
    int tx = threadIdx.x, ty = threadIdx.y;
#pragma unroll
    for (int j = 0; j < 4; ++j)
        tile[ty + j * 8][tx] = W[(size_t)(k0 + ty + j * 8) * N + n0 + tx];
    __syncthreads();
#pragma unroll
    for (int j = 0; j < 4; ++j)
        Wt[(size_t)(n0 + ty + j * 8) * K + k0 + tx] = f2bf(tile[tx][ty + j * 8]);
}

// ---------------------------------------------------------------------------
// bf16 WMMA GEMM: C[M,N] = A[M,K] * Bt[N,K]^T. Both operands bf16, K-contiguous;
// tiles staged with global_load_async_to_lds_b128, double-buffered (ASYNCcnt).
// Block tile 128x128, K-step 32, 256 threads = 8 waves (4x2, 32x64 each).
// mode 0: fp32 C row-major.   mode 1: bf16 head-major [b][h][s][d].
// mode 2: bf16 d-major        [b][h][d][s]  (for V -> attention B-operand).
// ---------------------------------------------------------------------------
__device__ __forceinline__ void gemm_stage_async(
    const unsigned short* __restrict__ Ab, const unsigned short* __restrict__ Bt,
    int m0, int n0, int K, int kk, int tid, unsigned sa, unsigned sb)
{
#pragma unroll
    for (int it = 0; it < 2; ++it) {           // A tile: 128 rows x 64B
        int idx = tid + it * 256;
        int row = idx >> 2, c = idx & 3;
        async_ld_b128(sa + row * 64 + c * 16, Ab + (size_t)(m0 + row) * K + kk + c * 8);
    }
#pragma unroll
    for (int it = 0; it < 2; ++it) {           // B tile: 128 rows x 64B
        int idx = tid + it * 256;
        int row = idx >> 2, c = idx & 3;
        async_ld_b128(sb + row * 64 + c * 16, Bt + (size_t)(n0 + row) * K + kk + c * 8);
    }
}

__global__ __launch_bounds__(256, 1)
void gemm_bf16wmma(const unsigned short* __restrict__ Ab,
                   const unsigned short* __restrict__ Bt,
                   float* __restrict__ C, unsigned short* __restrict__ Cb,
                   int M, int N, int K, int mode, int heads)
{
    __shared__ unsigned short s_a[2][128 * 32];   // [buf][row][k]
    __shared__ unsigned short s_b[2][128 * 32];   // [buf][n][k]

    const int tid  = threadIdx.x;
    const int lane = tid & 31;
    const int w    = tid >> 5;
    const int wm   = w >> 1;
    const int wn   = w & 1;
    const int hi   = lane >> 4;
    const int ln15 = lane & 15;
    const int m0   = blockIdx.x * 128;
    const int n0   = blockIdx.y * 128;

    const unsigned sa0 = lds_addr(&s_a[0][0]);
    const unsigned sb0 = lds_addr(&s_b[0][0]);

    v8f acc[2][4];
    const v8f vzero = {0.f, 0.f, 0.f, 0.f, 0.f, 0.f, 0.f, 0.f};
#pragma unroll
    for (int mt = 0; mt < 2; ++mt)
#pragma unroll
        for (int nt = 0; nt < 4; ++nt) acc[mt][nt] = vzero;

    const int nk = K >> 5;
    gemm_stage_async(Ab, Bt, m0, n0, K, 0, tid, sa0, sb0);     // prologue: tile 0 -> buf 0

    for (int t = 0; t < nk; ++t) {
        const int buf = t & 1;
        if (t + 1 < nk) {
            gemm_stage_async(Ab, Bt, m0, n0, K, (t + 1) << 5, tid,
                             sa0 + ((t + 1) & 1) * 8192, sb0 + ((t + 1) & 1) * 8192);
            asm volatile("s_wait_asynccnt 4" ::: "memory");    // tile t landed; t+1 in flight
        } else {
            asm volatile("s_wait_asynccnt 0" ::: "memory");
        }
        __syncthreads();

        const unsigned int* au = (const unsigned int*)s_a[buf];
        const unsigned int* bu = (const unsigned int*)s_b[buf];
        Frag afr[2], bfr[4];
#pragma unroll
        for (int mt = 0; mt < 2; ++mt) {
            int mrow = wm * 32 + mt * 16 + ln15;
#pragma unroll
            for (int i = 0; i < 8; ++i) {
                afr[mt].u[i] = au[mrow * 16 + i + hi * 8];
            }
        }
#pragma unroll
        for (int nt = 0; nt < 4; ++nt) {
            int n = wn * 64 + nt * 16 + ln15;
#pragma unroll
            for (int i = 0; i < 8; ++i)
                bfr[nt].u[i] = bu[n * 16 + i + hi * 8];            // B: K pairs contiguous
        }
#pragma unroll
        for (int mt = 0; mt < 2; ++mt)
#pragma unroll
            for (int nt = 0; nt < 4; ++nt)
                acc[mt][nt] = __builtin_amdgcn_wmma_f32_16x16x32_bf16(
                    false, afr[mt].v, false, bfr[nt].v, (short)0, acc[mt][nt],
                    false, false);
        __syncthreads();
    }

#pragma unroll
    for (int mt = 0; mt < 2; ++mt)
#pragma unroll
        for (int nt = 0; nt < 4; ++nt)
#pragma unroll
            for (int r = 0; r < 8; ++r) {
                int row = m0 + wm * 32 + mt * 16 + r + hi * 8;
                int col = n0 + wn * 64 + nt * 16 + ln15;
                float val = acc[mt][nt][r];
                if (mode == 0) {
                    C[(size_t)row * N + col] = val;
                } else {
                    int b = row >> 11, s = row & 2047;
                    int hh = col >> 8, d = col & 255;
                    if (mode == 1)
                        Cb[(((size_t)(b * heads) + hh) * 2048 + s) * 256 + d] = f2bf(val);
                    else
                        Cb[(((size_t)(b * heads) + hh) * 256 + d) * 2048 + s] = f2bf(val);
                }
            }
}

// ---------------------------------------------------------------------------
// In-place RoPE on bf16 head-major [b][h][s][256]
// ---------------------------------------------------------------------------
__global__ void rope_inplace(unsigned short* __restrict__ Q,
                             const float* __restrict__ cs,
                             const float* __restrict__ sn, int heads)
{
    int idx = blockIdx.x * blockDim.x + threadIdx.x;
    int total = 2 * 2048 * heads * 128;
    if (idx >= total) return;
    int j = idx & 127;
    int t = idx >> 7;
    int h = t % heads;  t /= heads;
    int s = t & 2047;
    int b = t >> 11;
    unsigned short* row = Q + (((size_t)(b * heads + h)) * 2048 + s) * 256;
    float x1 = bf2f(row[j]), x2 = bf2f(row[j + 128]);
    float c = cs[s * 128 + j], si = sn[s * 128 + j];
    row[j]       = f2bf(x1 * c - x2 * si);
    row[j + 128] = f2bf(x1 * si + x2 * c);
}

// ---------------------------------------------------------------------------
// Flash attention (GQA g=2, D=256, causal + window 1024, tanh softcap 50).
// Grid (S/64, B*N_HEADS), 128 threads = 4 waves, each wave owns 16 q-rows.
// Dynamic LDS, double-buffered async staging: K/V tiles for t+1 are in flight
// (ASYNCcnt) while tile t is consumed. Layout (69632 B):
//   [0,     32K)  s_k  x2  [t][d]   (16KB each)
//   [32K,   64K)  s_vt x2  [d][t]   (16KB each)
//   [64K,   68K)  s_p  x4 waves     (1KB each)
// ---------------------------------------------------------------------------
__device__ __forceinline__ void attn_stage_async(
    const unsigned short* __restrict__ gk, const unsigned short* __restrict__ gv,
    int t0, int tid, unsigned skb, unsigned svb)
{
#pragma unroll
    for (int it = 0; it < 8; ++it) {             // K tile: 32 rows x 512B
        int idx = tid + it * 128;
        int r = idx >> 5, c = idx & 31;
        async_ld_b128(skb + r * 512 + c * 16, gk + (size_t)(t0 + r) * 256 + c * 8);
    }
#pragma unroll
    for (int it = 0; it < 8; ++it) {             // V tile: 256 rows x 64B (d-major)
        int idx = tid + it * 128;
        int d = idx >> 2, c = idx & 3;
        async_ld_b128(svb + d * 64 + c * 16, gv + (size_t)d * 2048 + t0 + c * 8);
    }
}

__global__ __launch_bounds__(128, 1)
void attn_fa(const unsigned short* __restrict__ Qh,
             const unsigned short* __restrict__ Kh,
             const unsigned short* __restrict__ Vt,
             unsigned short* __restrict__ AoutB)
{
    extern __shared__ char smem[];

    const int tid  = threadIdx.x;
    const int lane = tid & 31;
    const int w    = tid >> 5;
    const int hi   = lane >> 4;
    const int ln15 = lane & 15;

    const int qblk = blockIdx.x * 64;
    const int bh   = blockIdx.y;               // b*16 + h
    const int b    = bh >> 4, h = bh & 15, kh = h >> 1;
    const int q0   = qblk + w * 16;

    const unsigned int* qrow =
        (const unsigned int*)Qh + (size_t)(((b * 16 + h) * 2048) + q0 + ln15) * 128;
    const unsigned short* gk = Kh + (size_t)((b * 8 + kh) * 2048) * 256;
    const unsigned short* gv = Vt + (size_t)((b * 8 + kh) * 256) * 2048;

    const unsigned sk0   = lds_addr(smem);              // + buf*16384
    const unsigned svt0  = lds_addr(smem + 32768);      // + buf*16384
    unsigned short* s_pw = (unsigned short*)(smem + 65536) + w * 512;

    v8f o[16];
    const v8f vzero = {0.f, 0.f, 0.f, 0.f, 0.f, 0.f, 0.f, 0.f};
#pragma unroll
    for (int dt = 0; dt < 16; ++dt) o[dt] = vzero;
    float m_r[8], l_r[8];
#pragma unroll
    for (int r = 0; r < 8; ++r) { m_r[r] = -1e30f; l_r[r] = 0.f; }

    int ts = qblk - 1023; if (ts < 0) ts = 0; ts &= ~31;
    const int niter = (qblk + 64 - ts) >> 5;

    attn_stage_async(gk, gv, ts, tid, sk0, svt0);       // prologue -> buf 0

    for (int i = 0; i < niter; ++i) {
        const int t0  = ts + (i << 5);
        const int buf = i & 1;
        if (i + 1 < niter) {
            attn_stage_async(gk, gv, t0 + 32, tid,
                             sk0 + ((i + 1) & 1) * 16384, svt0 + ((i + 1) & 1) * 16384);
            asm volatile("s_wait_asynccnt 16" ::: "memory");  // tile i landed; i+1 in flight
        } else {
            asm volatile("s_wait_asynccnt 0" ::: "memory");
        }
        __syncthreads();

        const unsigned int* sku = (const unsigned int*)(smem + buf * 16384);
        const unsigned int* vtu = (const unsigned int*)(smem + 32768 + buf * 16384);

        // --- S = Q K^T : 8 d-chunks x 2 n-tiles ---
        v8f sacc[2];
        sacc[0] = vzero; sacc[1] = vzero;
#pragma unroll
        for (int c = 0; c < 8; ++c) {
            Frag qa;
#pragma unroll
            for (int i2 = 0; i2 < 8; ++i2) {
                qa.u[i2] = qrow[c * 16 + i2 + hi * 8];
            }
#pragma unroll
            for (int nt = 0; nt < 2; ++nt) {
                Frag kb;
                int trow = nt * 16 + ln15;
#pragma unroll
                for (int i2 = 0; i2 < 8; ++i2)
                    kb.u[i2] = sku[trow * 128 + c * 16 + i2 + hi * 8];
                sacc[nt] = __builtin_amdgcn_wmma_f32_16x16x32_bf16(
                    false, qa.v, false, kb.v, (short)0, sacc[nt], false, false);
            }
        }

        // --- scale, softcap, mask ---
        float sv[2][8];
#pragma unroll
        for (int nt = 0; nt < 2; ++nt)
#pragma unroll
            for (int r = 0; r < 8; ++r) {
                float x = sacc[nt][r] * 0.0625f;            // 1/sqrt(256)
                x = fast_tanh(x * (1.0f / 50.0f)) * 50.0f;  // softcap
                int row  = q0 + r + hi * 8;
                int tcol = t0 + nt * 16 + ln15;
                if (tcol > row || (row - tcol) >= 1024) x = NEG_BIG;
                sv[nt][r] = x;
            }

        // --- online softmax (row stats via 16-lane-half shfl) ---
        float alph[8];
#pragma unroll
        for (int r = 0; r < 8; ++r) {
            float v = fmaxf(sv[0][r], sv[1][r]);
#pragma unroll
            for (int off = 1; off < 16; off <<= 1)
                v = fmaxf(v, __shfl_xor(v, off, 32));
            float mnew  = fmaxf(m_r[r], v);
            float alpha = __expf(m_r[r] - mnew);
            float p0 = __expf(sv[0][r] - mnew);
            float p1 = __expf(sv[1][r] - mnew);
            sv[0][r] = p0; sv[1][r] = p1;
            float s2 = p0 + p1;
#pragma unroll
            for (int off = 1; off < 16; off <<= 1)
                s2 += __shfl_xor(s2, off, 32);
            l_r[r] = l_r[r] * alpha + s2;
            m_r[r] = mnew;
            alph[r] = alpha;
        }
#pragma unroll
        for (int dt = 0; dt < 16; ++dt)
#pragma unroll
            for (int r = 0; r < 8; ++r) o[dt][r] *= alph[r];

        // --- P -> LDS (C-layout to A-layout within the wave) ---
#pragma unroll
        for (int nt = 0; nt < 2; ++nt)
#pragma unroll
            for (int r = 0; r < 8; ++r)
                s_pw[(r + hi * 8) * 32 + nt * 16 + ln15] = f2bf(sv[nt][r]);
        asm volatile("s_wait_dscnt 0" ::: "memory");

        Frag pa;
        {
            const unsigned int* pu = (const unsigned int*)s_pw;
#pragma unroll
            for (int i2 = 0; i2 < 8; ++i2) {
                pa.u[i2] = pu[ln15 * 16 + i2 + hi * 8];
            }
        }

        // --- O += P * V : 16 d-tiles ---
#pragma unroll
        for (int dt = 0; dt < 16; ++dt) {
            Frag vb;
            int n = dt * 16 + ln15;
#pragma unroll
            for (int i2 = 0; i2 < 8; ++i2)
                vb.u[i2] = vtu[n * 16 + i2 + hi * 8];
            o[dt] = __builtin_amdgcn_wmma_f32_16x16x32_bf16(
                false, pa.v, false, vb.v, (short)0, o[dt], false, false);
        }
        __syncthreads();
    }

    // --- finalize: O * (1/l), write bf16 [b*2048+s][h*256+d] ---
    float linv[8];
#pragma unroll
    for (int r = 0; r < 8; ++r) linv[r] = 1.0f / l_r[r];
#pragma unroll
    for (int dt = 0; dt < 16; ++dt)
#pragma unroll
        for (int r = 0; r < 8; ++r) {
            int row = q0 + r + hi * 8;
            int col = h * 256 + dt * 16 + ln15;
            AoutB[(size_t)(b * 2048 + row) * 4096 + col] = f2bf(o[dt][r] * linv[r]);
        }
}

// ---------------------------------------------------------------------------
// Workspace layout (bytes, all bf16):
//   [0,          28.0M)  Xb    [4096][3584]
//   [28.0M,      56.0M)  Wqt   [4096][3584]
//   [56.0M,      70.0M)  Wkt   [2048][3584]
//   [70.0M,      84.0M)  Wvt   [2048][3584]
//   [84.0M,     112.0M)  Wot   [3584][4096]
//   [112.0M,    144.0M)  Qh    [b][16][2048][256]   (RoPE in place)
//   [144.0M,    160.0M)  Kh    [b][8][2048][256]    (RoPE in place)
//   [160.0M,    176.0M)  Vt    [b][8][256][2048]
//   [176.0M,    208.0M)  attnB [4096][4096]
// ---------------------------------------------------------------------------
extern "C" void kernel_launch(void* const* d_in, const int* in_sizes, int n_in,
                              void* d_out, int out_size, void* d_ws, size_t ws_size,
                              hipStream_t stream) {
    (void)in_sizes; (void)n_in; (void)out_size; (void)ws_size;
    const float* X  = (const float*)d_in[0];
    const float* cs = (const float*)d_in[1];
    const float* sn = (const float*)d_in[2];
    const float* wq = (const float*)d_in[4];
    const float* wk = (const float*)d_in[5];
    const float* wv = (const float*)d_in[6];
    const float* wo = (const float*)d_in[7];

    char* ws = (char*)d_ws;
    unsigned short* Xb   = (unsigned short*)(ws);
    unsigned short* Wqt  = (unsigned short*)(ws + 29360128);
    unsigned short* Wkt  = (unsigned short*)(ws + 58720256);
    unsigned short* Wvt  = (unsigned short*)(ws + 73400320);
    unsigned short* Wot  = (unsigned short*)(ws + 88080384);
    unsigned short* Qh   = (unsigned short*)(ws + 117440512);
    unsigned short* Kh   = (unsigned short*)(ws + 150994944);
    unsigned short* Vt   = (unsigned short*)(ws + 167772160);
    unsigned short* attnB= (unsigned short*)(ws + 184549376);

    // operand conversion / transposition (bf16)
    int n4x = (4096 * 3584) / 4;
    cvt_bf16_vec4<<<(n4x + 255) / 256, 256, 0, stream>>>((const float4*)X, (uint2*)Xb, n4x);
    transpose_cvt<<<dim3(128, 112), dim3(32, 8), 0, stream>>>(wq, Wqt, 3584, 4096);
    transpose_cvt<<<dim3(64, 112),  dim3(32, 8), 0, stream>>>(wk, Wkt, 3584, 2048);
    transpose_cvt<<<dim3(64, 112),  dim3(32, 8), 0, stream>>>(wv, Wvt, 3584, 2048);
    transpose_cvt<<<dim3(112, 128), dim3(32, 8), 0, stream>>>(wo, Wot, 4096, 3584);

    // QKV projections (async double-buffered bf16 WMMA GEMM)
    gemm_bf16wmma<<<dim3(32, 32), 256, 0, stream>>>(Xb, Wqt, nullptr, Qh, 4096, 4096, 3584, 1, 16);
    gemm_bf16wmma<<<dim3(32, 16), 256, 0, stream>>>(Xb, Wkt, nullptr, Kh, 4096, 2048, 3584, 1, 8);
    gemm_bf16wmma<<<dim3(32, 16), 256, 0, stream>>>(Xb, Wvt, nullptr, Vt, 4096, 2048, 3584, 2, 8);

    // RoPE in place
    int tq = 2 * 2048 * 16 * 128;
    rope_inplace<<<(tq + 255) / 256, 256, 0, stream>>>(Qh, cs, sn, 16);
    int tk = 2 * 2048 * 8 * 128;
    rope_inplace<<<(tk + 255) / 256, 256, 0, stream>>>(Kh, cs, sn, 8);

    // Flash attention (dynamic LDS: 69632 B) -> bf16
    attn_fa<<<dim3(32, 32), 128, 69632, stream>>>(Qh, Kh, Vt, attnB);

    // Output projection -> fp32 d_out
    gemm_bf16wmma<<<dim3(32, 28), 256, 0, stream>>>(attnB, Wot, (float*)d_out, nullptr,
                                                    4096, 3584, 4096, 0, 0);
}
